// CricketHeteroGNN_5428838662520
// MI455X (gfx1250) — matrix operand order repaired and checked by reading, not compiled, not run on hardware
//
#include <hip/hip_runtime.h>
#include <stdint.h>

#define H        128
#define N_BALL   200000
#define N_QUERY  8192
#define N_PLAYER 180224
#define N_VENUE  8192
#define N_TEAM   16384
#define E_BB     400000
#define E_PB     400000
#define E_BQ     500000

typedef __bf16 bf16_t;
typedef bf16_t v16bf __attribute__((ext_vector_type(16)));
typedef float  v8f   __attribute__((ext_vector_type(8)));

union ABFrag { v16bf v; uint4 q[2]; };
union Pack8  { unsigned short u[8]; uint4 q; };

__device__ __forceinline__ unsigned short f32_to_bf16_bits(float f) {
  unsigned int u = __float_as_uint(f);
  u += 0x7FFFu + ((u >> 16) & 1u);           // round-to-nearest-even
  return (unsigned short)(u >> 16);
}
__device__ __forceinline__ float bf16_bits_to_f32(unsigned short h) {
  return __uint_as_float(((unsigned int)h) << 16);
}

// ---- weight prep: f32 [K][N] -> bf16 transposed, packed at dst[n*dstStride + dstOff + k] ----
__global__ void transpose_pack_bf16_kernel(const float* __restrict__ src,
                                           unsigned short* __restrict__ dst,
                                           int K, int N, int dstStride, int dstOff) {
  int idx = blockIdx.x * blockDim.x + threadIdx.x;
  if (idx < K * N) {
    int k = idx / N, n = idx % N;
    dst[(size_t)n * dstStride + dstOff + k] = f32_to_bf16_bits(src[idx]);
  }
}

// ---- degree (segment counts, constant across layers) ----
__global__ void deg_kernel(const int* __restrict__ dst, int n, float* __restrict__ deg) {
  int i = blockIdx.x * blockDim.x + threadIdx.x;
  if (i < n) atomicAdd(&deg[dst[i]], 1.0f);
}

// ---- encoders (small K, VALU; memory-trivial) ----
template<int K>
__global__ void enc_kernel(const float* __restrict__ table, const int* __restrict__ ids,
                           const float* __restrict__ W, const float* __restrict__ bias,
                           unsigned short* __restrict__ out, int n) {
  __shared__ float sW[K * H];
  for (int i = threadIdx.x; i < K * H; i += blockDim.x) sW[i] = W[i];
  __syncthreads();
  int col = threadIdx.x & (H - 1);
  int sub = threadIdx.x >> 7;
  int base = blockIdx.x * 16;
  for (int i = sub; i < 16; i += 2) {
    int node = base + i;
    if (node >= n) continue;
    int srow = ids ? ids[node] : node;
    const float* row = table + (size_t)srow * K;
    float acc = bias[col];
#pragma unroll
    for (int k = 0; k < K; ++k) acc = fmaf(row[k], sW[k * H + col], acc);
    out[(size_t)node * H + col] = f32_to_bf16_bits(acc);
  }
}

__global__ void enc_player_kernel(const float* __restrict__ ptab, const float* __restrict__ rtab,
                                  const int* __restrict__ pid, const int* __restrict__ rid,
                                  const float* __restrict__ W, const float* __restrict__ bias,
                                  unsigned short* __restrict__ out, int n) {
  __shared__ float sW[80 * H];
  for (int i = threadIdx.x; i < 80 * H; i += blockDim.x) sW[i] = W[i];
  __syncthreads();
  int col = threadIdx.x & (H - 1);
  int sub = threadIdx.x >> 7;
  int base = blockIdx.x * 16;
  for (int i = sub; i < 16; i += 2) {
    int node = base + i;
    if (node >= n) continue;
    const float* prow = ptab + (size_t)pid[node] * 64;
    const float* rrow = rtab + (size_t)rid[node] * 16;
    float acc = bias[col];
#pragma unroll
    for (int k = 0; k < 64; ++k) acc = fmaf(prow[k], sW[k * H + col], acc);
#pragma unroll
    for (int k = 0; k < 16; ++k) acc = fmaf(rrow[k], sW[(64 + k) * H + col], acc);
    out[(size_t)node * H + col] = f32_to_bf16_bits(acc);
  }
}

// ---- feature scatter: agg[dst] += x[src] / deg[dst]  (seg_mean commutes with @W) ----
__global__ void __launch_bounds__(256)
edge_scatter_kernel(const unsigned short* __restrict__ xsrc,
                    const int* __restrict__ src, const int* __restrict__ dst,
                    const float* __restrict__ deg,
                    float* __restrict__ agg, int n_edges) {
  int tid  = threadIdx.x;
  int esub = tid >> 5;                 // 8 edges per block
  int lane = tid & 31;                 // 4 columns per lane
  int e = blockIdx.x * 8 + esub;
  if (e >= n_edges) return;
  int s = src[e], d = dst[e];
  float scale = 1.0f / fmaxf(deg[d], 1.0f);
  const unsigned short* xr = xsrc + (size_t)s * H + lane * 4;
  float* ar = agg + (size_t)d * H + lane * 4;
  uint2 pk = *(const uint2*)xr;        // 4 bf16
  atomicAdd(&ar[0], bf16_bits_to_f32((unsigned short)(pk.x & 0xffffu)) * scale);
  atomicAdd(&ar[1], bf16_bits_to_f32((unsigned short)(pk.x >> 16)) * scale);
  atomicAdd(&ar[2], bf16_bits_to_f32((unsigned short)(pk.y & 0xffffu)) * scale);
  atomicAdd(&ar[3], bf16_bits_to_f32((unsigned short)(pk.y >> 16)) * scale);
}

// ---- WMMA tile core: 16 rows x 16 cols, K = KT, A from LDS, B = packed W^T [128][KT] ----
template<int KT>
__device__ __forceinline__ v8f wmma_row_tile(const unsigned short* __restrict__ sA,
                                             const unsigned short* __restrict__ Wt,
                                             int lane, int n0) {
  int hf = lane >> 4;       // K-half selector
  int mlow = lane & 15;     // A row / B,D column-in-tile
  ABFrag a, b;
  v8f c = {};
  const unsigned short* arow = sA + mlow * KT;
  const unsigned short* wcol = Wt + (size_t)(n0 + mlow) * KT;
#pragma unroll
  for (int k0 = 0; k0 < KT; k0 += 32) {
    const uint4* rp = (const uint4*)arow;
    a.q[0] = rp[(k0 >> 3) + hf];          // K in [k0+hf*8, +8)
    a.q[1] = rp[(k0 >> 3) + 2 + hf];      // K in [k0+16+hf*8, +8)
    const uint4* bp = (const uint4*)(wcol + k0 + (hf << 3));
    b.q[0] = bp[0];
    b.q[1] = bp[2];
    c = __builtin_amdgcn_wmma_f32_16x16x32_bf16(false, a.v, false, b.v,
                                                (short)0, c, false, false);
  }
  return c;
}

// ---- fused conv block: x = LN(relu([x|agg0|agg1|(agg2)] @ Wcat^T + b)) in place ----
// KTOT = 128 * (1 + #relations); n_nodes multiple of 16.
template<int KTOT>
__global__ void __launch_bounds__(256)
combine_kernel(unsigned short* __restrict__ x,
               const unsigned short* __restrict__ Wt,
               const float* __restrict__ bias,
               const float* __restrict__ agg0,
               const float* __restrict__ agg1,
               const float* __restrict__ agg2,
               const float* __restrict__ g, const float* __restrict__ bv) {
  __shared__ __align__(16) unsigned short sA[16 * KTOT];
  __shared__ float sOut[16 * H];
  __shared__ float sMean[16], sRstd[16];
  constexpr int CH = KTOT / 8;            // 8-element chunks per row
  int row_base = blockIdx.x * 16;
  int tid = threadIdx.x;
  for (int idx = tid; idx < 16 * CH; idx += 256) {
    int r = idx / CH, ch = idx % CH;
    int seg = ch >> 4, chin = ch & 15;
    if (seg == 0) {
      ((uint4*)(sA + r * KTOT))[ch] =
          ((const uint4*)(x + (size_t)(row_base + r) * H))[chin];
    } else {
      const float* ag = (seg == 1) ? agg0 : (seg == 2) ? agg1 : agg2;
      const float* p = ag + (size_t)(row_base + r) * H + chin * 8;
      Pack8 t;
#pragma unroll
      for (int j = 0; j < 8; ++j) t.u[j] = f32_to_bf16_bits(p[j]);
      ((uint4*)(sA + r * KTOT))[ch] = t.q;
    }
  }
  __syncthreads();

  int lane = tid & 31;
  int n0 = (tid >> 5) << 4;
  v8f c = wmma_row_tile<KTOT>(sA, Wt, lane, n0);

  int hf = lane >> 4;
  int colN = n0 + (lane & 15);
#pragma unroll
  for (int r = 0; r < 8; ++r) {
    int m = (hf << 3) + r;
    sOut[m * H + colN] = fmaxf(c[r] + bias[colN], 0.0f);
  }
  __syncthreads();
  if (tid < 16) {
    float s = 0.f;
    for (int j = 0; j < H; ++j) s += sOut[tid * H + j];
    float mean = s * (1.0f / H);
    float var = 0.f;
    for (int j = 0; j < H; ++j) { float d = sOut[tid * H + j] - mean; var += d * d; }
    sMean[tid] = mean;
    sRstd[tid] = rsqrtf(var * (1.0f / H) + 1e-5f);
  }
  __syncthreads();
  for (int idx = tid; idx < 16 * H; idx += 256) {
    int r = idx >> 7, col = idx & (H - 1);
    float v = (sOut[idx] - sMean[r]) * sRstd[r] * g[col] + bv[col];
    x[(size_t)(row_base + r) * H + col] = f32_to_bf16_bits(v);
  }
}

// ---- predictor layer: out = gelu(LN(x @ W + b)), NOUT in {128, 64} ----
template<int NOUT>
__global__ void __launch_bounds__(NOUT * 2)
mlp_kernel(const unsigned short* __restrict__ xin,
           const unsigned short* __restrict__ Wt,
           const float* __restrict__ bias,
           const float* __restrict__ g, const float* __restrict__ bv,
           unsigned short* __restrict__ xout) {
  __shared__ __align__(16) unsigned short sA[16 * H];
  __shared__ float sOut[16 * NOUT];
  __shared__ float sMean[16], sRstd[16];
  const int BT = NOUT * 2;
  int row_base = blockIdx.x * 16;
  int tid = threadIdx.x;
  for (int idx = tid; idx < 256; idx += BT) {
    int r = idx >> 4, ch = idx & 15;
    ((uint4*)sA)[idx] = ((const uint4*)(xin + (size_t)(row_base + r) * H))[ch];
  }
  __syncthreads();

  int lane = tid & 31;
  int n0 = (tid >> 5) << 4;
  v8f c = wmma_row_tile<H>(sA, Wt, lane, n0);

  int hf = lane >> 4;
  int colN = n0 + (lane & 15);
#pragma unroll
  for (int r = 0; r < 8; ++r) {
    int m = (hf << 3) + r;
    sOut[m * NOUT + colN] = c[r] + bias[colN];
  }
  __syncthreads();
  if (tid < 16) {
    float s = 0.f;
    for (int j = 0; j < NOUT; ++j) s += sOut[tid * NOUT + j];
    float mean = s * (1.0f / NOUT);
    float var = 0.f;
    for (int j = 0; j < NOUT; ++j) { float d = sOut[tid * NOUT + j] - mean; var += d * d; }
    sMean[tid] = mean;
    sRstd[tid] = rsqrtf(var * (1.0f / NOUT) + 1e-5f);
  }
  __syncthreads();
  for (int idx = tid; idx < 16 * NOUT; idx += BT) {
    int r = idx / NOUT, col = idx % NOUT;
    float hh = (sOut[idx] - sMean[r]) * sRstd[r] * g[col] + bv[col];
    float t = 0.7978845608028654f * (hh + 0.044715f * hh * hh * hh);
    float gel = 0.5f * hh * (1.0f + tanhf(t));
    xout[(size_t)(row_base + r) * NOUT + col] = f32_to_bf16_bits(gel);
  }
}

// ---- final tiny GEMM: logits[8192,7] ----
__global__ void pred3_kernel(const unsigned short* __restrict__ h2,
                             const float* __restrict__ W3, const float* __restrict__ b3,
                             float* __restrict__ out, int n) {
  int idx = blockIdx.x * blockDim.x + threadIdx.x;
  if (idx >= n * 7) return;
  int q = idx / 7, j = idx % 7;
  float acc = b3[j];
  const unsigned short* hr = h2 + (size_t)q * 64;
#pragma unroll
  for (int k = 0; k < 64; ++k) acc = fmaf(bf16_bits_to_f32(hr[k]), W3[k * 7 + j], acc);
  out[idx] = acc;
}

extern "C" void kernel_launch(void* const* d_in, const int* in_sizes, int n_in,
                              void* d_out, int out_size, void* d_ws, size_t ws_size,
                              hipStream_t stream) {
  (void)in_sizes; (void)n_in; (void)out_size; (void)ws_size;
  const float* ball_x       = (const float*)d_in[0];
  const float* query_x      = (const float*)d_in[1];
  const float* player_table = (const float*)d_in[2];
  const float* role_table   = (const float*)d_in[3];
  const float* venue_table  = (const float*)d_in[4];
  const float* team_table   = (const float*)d_in[5];
  const float* enc_W_venue  = (const float*)d_in[6];
  const float* enc_b_venue  = (const float*)d_in[7];
  const float* enc_W_team   = (const float*)d_in[8];
  const float* enc_b_team   = (const float*)d_in[9];
  const float* enc_W_player = (const float*)d_in[10];
  const float* enc_b_player = (const float*)d_in[11];
  const float* enc_W_ball   = (const float*)d_in[12];
  const float* enc_b_ball   = (const float*)d_in[13];
  const float* enc_W_query  = (const float*)d_in[14];
  const float* enc_b_query  = (const float*)d_in[15];
  const float* conv_rel_W   = (const float*)d_in[16];
  const float* conv_self_W  = (const float*)d_in[17];
  const float* conv_self_b  = (const float*)d_in[18];
  const float* ln_g         = (const float*)d_in[19];
  const float* ln_b         = (const float*)d_in[20];
  const float* pred_W1      = (const float*)d_in[21];
  const float* pred_b1      = (const float*)d_in[22];
  const float* pred_g1      = (const float*)d_in[23];
  const float* pred_be1     = (const float*)d_in[24];
  const float* pred_W2      = (const float*)d_in[25];
  const float* pred_b2      = (const float*)d_in[26];
  const float* pred_g2      = (const float*)d_in[27];
  const float* pred_be2     = (const float*)d_in[28];
  const float* pred_W3      = (const float*)d_in[29];
  const float* pred_b3      = (const float*)d_in[30];
  const int* venue_id  = (const int*)d_in[31];
  const int* team_id   = (const int*)d_in[32];
  const int* player_id = (const int*)d_in[33];
  const int* role_id   = (const int*)d_in[34];
  const int* bb_src = (const int*)d_in[35];
  const int* bb_dst = (const int*)d_in[36];
  const int* pb_src = (const int*)d_in[37];
  const int* pb_dst = (const int*)d_in[38];
  const int* bq_src = (const int*)d_in[39];
  const int* bq_dst = (const int*)d_in[40];
  const int* vq_src = (const int*)d_in[41];
  const int* vq_dst = (const int*)d_in[42];
  const int* tq_src = (const int*)d_in[43];
  const int* tq_dst = (const int*)d_in[44];

  // ---- workspace carve-up ----
  char* ws = (char*)d_ws;
  size_t off = 0;
  auto take = [&](size_t bytes) -> char* {
    char* p = ws + off;
    off = (off + bytes + 255) & ~(size_t)255;
    return p;
  };
  // packed per-layer concat weights: ball [128][384], query [128][512]
  unsigned short* wcat_b[3];
  unsigned short* wcat_q[3];
  for (int l = 0; l < 3; ++l) wcat_b[l] = (unsigned short*)take((size_t)H * 384 * 2);
  for (int l = 0; l < 3; ++l) wcat_q[l] = (unsigned short*)take((size_t)H * 512 * 2);
  unsigned short* wt_p1 = (unsigned short*)take((size_t)H * H * 2);
  unsigned short* wt_p2 = (unsigned short*)take((size_t)64 * H * 2);
  unsigned short* xb = (unsigned short*)take((size_t)N_BALL * H * 2);
  unsigned short* xp = (unsigned short*)take((size_t)N_PLAYER * H * 2);
  unsigned short* xq = (unsigned short*)take((size_t)N_QUERY * H * 2);
  unsigned short* xv = (unsigned short*)take((size_t)N_VENUE * H * 2);
  unsigned short* xt = (unsigned short*)take((size_t)N_TEAM * H * 2);
  unsigned short* h1 = (unsigned short*)take((size_t)N_QUERY * H * 2);
  unsigned short* h2 = (unsigned short*)take((size_t)N_QUERY * 64 * 2);
  float* agg_bb = (float*)take((size_t)N_BALL * H * 4);
  float* agg_pb = (float*)take((size_t)N_BALL * H * 4);
  float* agg_bq = (float*)take((size_t)N_QUERY * H * 4);
  float* agg_vq = (float*)take((size_t)N_QUERY * H * 4);
  float* agg_tq = (float*)take((size_t)N_QUERY * H * 4);
  float* deg_bb = (float*)take((size_t)N_BALL * 4);
  float* deg_pb = (float*)take((size_t)N_BALL * 4);
  float* deg_bq = (float*)take((size_t)N_QUERY * 4);
  float* deg_vq = (float*)take((size_t)N_QUERY * 4);
  float* deg_tq = (float*)take((size_t)N_QUERY * 4);

  // ---- weight prep: pack [Wself | Wrel...]^T per layer, bf16 ----
  const int TB = 256, TG = (H * H + 255) / 256;
  for (int l = 0; l < 3; ++l) {
    transpose_pack_bf16_kernel<<<TG, TB, 0, stream>>>(conv_self_W + (size_t)(l * 2 + 0) * H * H, wcat_b[l], H, H, 384, 0);
    transpose_pack_bf16_kernel<<<TG, TB, 0, stream>>>(conv_rel_W + (size_t)(l * 5 + 0) * H * H, wcat_b[l], H, H, 384, 128);
    transpose_pack_bf16_kernel<<<TG, TB, 0, stream>>>(conv_rel_W + (size_t)(l * 5 + 1) * H * H, wcat_b[l], H, H, 384, 256);
    transpose_pack_bf16_kernel<<<TG, TB, 0, stream>>>(conv_self_W + (size_t)(l * 2 + 1) * H * H, wcat_q[l], H, H, 512, 0);
    transpose_pack_bf16_kernel<<<TG, TB, 0, stream>>>(conv_rel_W + (size_t)(l * 5 + 2) * H * H, wcat_q[l], H, H, 512, 128);
    transpose_pack_bf16_kernel<<<TG, TB, 0, stream>>>(conv_rel_W + (size_t)(l * 5 + 3) * H * H, wcat_q[l], H, H, 512, 256);
    transpose_pack_bf16_kernel<<<TG, TB, 0, stream>>>(conv_rel_W + (size_t)(l * 5 + 4) * H * H, wcat_q[l], H, H, 512, 384);
  }
  transpose_pack_bf16_kernel<<<TG, TB, 0, stream>>>(pred_W1, wt_p1, H, H, H, 0);
  transpose_pack_bf16_kernel<<<(H * 64 + 255) / 256, TB, 0, stream>>>(pred_W2, wt_p2, H, 64, H, 0);

  // ---- degrees (constant across layers) ----
  hipMemsetAsync(deg_bb, 0, (size_t)N_BALL * 4, stream);
  hipMemsetAsync(deg_pb, 0, (size_t)N_BALL * 4, stream);
  hipMemsetAsync(deg_bq, 0, (size_t)N_QUERY * 4, stream);
  hipMemsetAsync(deg_vq, 0, (size_t)N_QUERY * 4, stream);
  hipMemsetAsync(deg_tq, 0, (size_t)N_QUERY * 4, stream);
  deg_kernel<<<(E_BB + 255) / 256, 256, 0, stream>>>(bb_dst, E_BB, deg_bb);
  deg_kernel<<<(E_PB + 255) / 256, 256, 0, stream>>>(pb_dst, E_PB, deg_pb);
  deg_kernel<<<(E_BQ + 255) / 256, 256, 0, stream>>>(bq_dst, E_BQ, deg_bq);
  deg_kernel<<<(N_VENUE + 255) / 256, 256, 0, stream>>>(vq_dst, N_VENUE, deg_vq);
  deg_kernel<<<(N_TEAM + 255) / 256, 256, 0, stream>>>(tq_dst, N_TEAM, deg_tq);

  // ---- encoders -> bf16 node features ----
  enc_kernel<16><<<(N_BALL + 15) / 16, 256, 0, stream>>>(ball_x, nullptr, enc_W_ball, enc_b_ball, xb, N_BALL);
  enc_kernel<16><<<(N_QUERY + 15) / 16, 256, 0, stream>>>(query_x, nullptr, enc_W_query, enc_b_query, xq, N_QUERY);
  enc_kernel<32><<<(N_VENUE + 15) / 16, 256, 0, stream>>>(venue_table, venue_id, enc_W_venue, enc_b_venue, xv, N_VENUE);
  enc_kernel<32><<<(N_TEAM + 15) / 16, 256, 0, stream>>>(team_table, team_id, enc_W_team, enc_b_team, xt, N_TEAM);
  enc_player_kernel<<<(N_PLAYER + 15) / 16, 256, 0, stream>>>(player_table, role_table, player_id, role_id,
                                                              enc_W_player, enc_b_player, xp, N_PLAYER);

  // ---- 3 hetero conv layers: scatter means, then fused concat-GEMM + ReLU + LN ----
  for (int l = 0; l < 3; ++l) {
    hipMemsetAsync(agg_bb, 0, (size_t)N_BALL * H * 4, stream);
    hipMemsetAsync(agg_pb, 0, (size_t)N_BALL * H * 4, stream);
    hipMemsetAsync(agg_bq, 0, (size_t)N_QUERY * H * 4, stream);
    hipMemsetAsync(agg_vq, 0, (size_t)N_QUERY * H * 4, stream);
    hipMemsetAsync(agg_tq, 0, (size_t)N_QUERY * H * 4, stream);
    edge_scatter_kernel<<<(E_BB + 7) / 8, 256, 0, stream>>>(xb, bb_src, bb_dst, deg_bb, agg_bb, E_BB);
    edge_scatter_kernel<<<(E_PB + 7) / 8, 256, 0, stream>>>(xp, pb_src, pb_dst, deg_pb, agg_pb, E_PB);
    edge_scatter_kernel<<<(E_BQ + 7) / 8, 256, 0, stream>>>(xb, bq_src, bq_dst, deg_bq, agg_bq, E_BQ);
    edge_scatter_kernel<<<(N_VENUE + 7) / 8, 256, 0, stream>>>(xv, vq_src, vq_dst, deg_vq, agg_vq, N_VENUE);
    edge_scatter_kernel<<<(N_TEAM + 7) / 8, 256, 0, stream>>>(xt, tq_src, tq_dst, deg_tq, agg_tq, N_TEAM);
    combine_kernel<384><<<N_BALL / 16, 256, 0, stream>>>(xb, wcat_b[l], conv_self_b + (l * 2 + 0) * H,
                                                         agg_bb, agg_pb, nullptr,
                                                         ln_g + (l * 2 + 0) * H, ln_b + (l * 2 + 0) * H);
    combine_kernel<512><<<N_QUERY / 16, 256, 0, stream>>>(xq, wcat_q[l], conv_self_b + (l * 2 + 1) * H,
                                                          agg_bq, agg_vq, agg_tq,
                                                          ln_g + (l * 2 + 1) * H, ln_b + (l * 2 + 1) * H);
  }

  // ---- predictor head ----
  mlp_kernel<128><<<N_QUERY / 16, 256, 0, stream>>>(xq, wt_p1, pred_b1, pred_g1, pred_be1, h1);
  mlp_kernel<64><<<N_QUERY / 16, 128, 0, stream>>>(h1, wt_p2, pred_b2, pred_g2, pred_be2, h2);
  pred3_kernel<<<(N_QUERY * 7 + 255) / 256, 256, 0, stream>>>(h2, pred_W3, pred_b3, (float*)d_out, N_QUERY);
}
